// CompoundLoss_20839181320289
// MI455X (gfx1250) — matrix operand order repaired and checked by reading, not compiled, not run on hardware
//
#include <hip/hip_runtime.h>
#include <hip/hip_bf16.h>
#include <math.h>

#define NC 11
#define IH 512
#define IW 512
#define NB 8
#define TILE 16

typedef __attribute__((ext_vector_type(16))) _Float16 v16h;
typedef __attribute__((ext_vector_type(8)))  float    v8f;
typedef __attribute__((ext_vector_type(4)))  unsigned int u32x4;
typedef __attribute__((ext_vector_type(8)))  int          i32x8;
typedef __attribute__((ext_vector_type(4)))  int          i32x4;

struct Accum {
  float inter[NB][NC];
  float tcnt [NB][NC];
  float psum [NB][NC];
  float focC [NB][NC];
  float num  [NB][NC];
  float den  [NB][NC];
};

__global__ void loss_init(float* w, int n) {
  int i = blockIdx.x * blockDim.x + threadIdx.x;
  if (i < n) w[i] = 0.0f;
}

// A-matrix (16x32 f16) K-slot mapping: lanes<16 hold K=0..15 ; lanes>=16 hold K=16..31
__device__ __forceinline__ int amapK(int j, bool lo) {
  return j + (lo ? 0 : 16);
}

// Gather B (32x16 f16) from a staged [pixel][16] f16 matrix: lane holds column n=lane&15,
// K slots j=0..15 map to pixels wbase + j (+16 for lanes>=16).
__device__ __forceinline__ v16h gatherB(const _Float16* buf, int wbase, int lane) {
  const int n   = lane & 15;
  const int off = (lane < 16) ? 0 : 16;
  v16h Bv;
  #pragma unroll
  for (int j = 0; j < 16; ++j)
    Bv[j] = buf[(wbase + off + j) * 16 + n];
  return Bv;
}

__global__ __launch_bounds__(256) void loss_main(const float* __restrict__ logits,
                                                 const int*   __restrict__ tgt,
                                                 Accum* __restrict__ acc)
{
  __shared__ int      sT[18 * 18];      // target tile + halo (-1 = OOB)
  __shared__ int      sCt[256];         // center target per pixel (compact)
  __shared__ float    sPt [256];        // raw pt per pixel
  __shared__ float    sFoc[256];        // focal term per pixel
  __shared__ _Float16 sMA[256 * 16];    // staged per-(pixel,class) matrix A (probs, then den)
  __shared__ _Float16 sMB[256 * 16];    // staged per-(pixel,class) matrix B (num)
  __shared__ float    sInter[16], sTc[16], sFc[16], sPs[16], sNum[16], sDen[16];

  const int tid  = threadIdx.x;
  const int bi   = blockIdx.x;          // 8 * 32 * 32 blocks
  const int b    = bi >> 10;
  const int tile = bi & 1023;
  const int ty   = (tile >> 5) * TILE;
  const int tx   = (tile & 31) * TILE;

  if (tid < 16) {
    sInter[tid] = 0.f; sTc[tid] = 0.f; sFc[tid] = 0.f;
    sPs[tid] = 0.f; sNum[tid] = 0.f; sDen[tid] = 0.f;
  }

  const int py = tid >> 4, px = tid & 15;
  const int gy = ty + py,  gx = tx + px;
  const size_t pixBase = (size_t)b * NC * IH * IW + (size_t)gy * IW + gx;

  __builtin_prefetch(&logits[pixBase], 0, 1);   // global_prefetch_b8

  // ---- TDM: core 16x16 target tile -> LDS at sT[1][1] with 2-dword row pad ----
  if (tid < 32) {
    const unsigned long long gaddr =
        (unsigned long long)(const void*)&tgt[(size_t)b * IH * IW + (size_t)ty * IW + tx];
    const unsigned int ldsAddr = (unsigned int)(unsigned long long)(void*)&sT[1 * 18 + 1];

    u32x4 g0;
    g0[0] = 1u;                                   // count=1 valid descriptor
    g0[1] = ldsAddr;                              // lds_addr
    g0[2] = (unsigned int)gaddr;                  // global_addr[31:0]
    g0[3] = (unsigned int)((gaddr >> 32) & 0x01FFFFFFull) | (2u << 30); // addr[56:32] | type=2

    const unsigned int TD = 1u << 20;             // generous tensor dims (tile always in-bounds)
    i32x8 g1;
    g1[0] = (int)((2u << 16)      // data_size = 4B
                | (1u << 20)      // pad_enable
                | (3u << 22)      // pad_interval: every 16 dwords
                | (1u << 25));    // pad_amount: 2 dwords
    g1[1] = (int)((TD & 0xFFFFu) << 16);          // tensor_dim0[15:0] @ bits 63:48
    g1[2] = (int)((TD >> 16) | ((TD & 0xFFFFu) << 16)); // tensor_dim0 hi | tensor_dim1 lo
    g1[3] = (int)((TD >> 16) | (16u << 16));      // tensor_dim1 hi | tile_dim0 = 16
    g1[4] = 16;                                   // tile_dim1 = 16 (tile_dim2 = 0)
    g1[5] = 512;                                  // tensor_dim0_stride = 512 dwords
    g1[6] = 0;
    g1[7] = 0;
    i32x4 gz4 = {0, 0, 0, 0};
    i32x8 gz8 = {0, 0, 0, 0, 0, 0, 0, 0};
    __builtin_amdgcn_tensor_load_to_lds(g0, g1, gz4, gz4, gz8, 0);
  }

  // ---- scalar halo: 68 perimeter texels of the 18x18 tile ----
  if (tid < 68) {
    int ly, lx;
    if      (tid < 18) { ly = 0;        lx = tid;      }
    else if (tid < 36) { ly = 17;       lx = tid - 18; }
    else if (tid < 52) { ly = tid - 35; lx = 0;        }
    else               { ly = tid - 51; lx = 17;       }
    int yy = ty + ly - 1, xx = tx + lx - 1;
    int v = -1;
    if (yy >= 0 && yy < IH && xx >= 0 && xx < IW)
      v = tgt[(size_t)b * IH * IW + (size_t)yy * IW + xx];
    sT[ly * 18 + lx] = v;
  }

  __builtin_amdgcn_s_wait_tensorcnt(0);
  __syncthreads();

  // ---- softmax + argmax over 11 classes (coalesced plane-strided loads) ----
  float lg[NC];
  lg[0] = logits[pixBase];
  float mx = lg[0];
  int   pred = 0;
  #pragma unroll
  for (int c = 1; c < NC; ++c) {
    lg[c] = logits[pixBase + (size_t)c * IH * IW];
    if (lg[c] > mx) { mx = lg[c]; pred = c; }
  }
  float se = 0.f;
  #pragma unroll
  for (int c = 0; c < NC; ++c) { lg[c] = __expf(lg[c] - mx); se += lg[c]; }
  const float inv = 1.0f / se;

  const int tc = sT[(py + 1) * 18 + (px + 1)];
  const float ptRaw = lg[tc] * inv;
  const float ptCl  = fminf(fmaxf(ptRaw, 1e-7f), 1.0f);
  const float om    = 1.0f - ptCl;
  sCt [tid] = tc;
  sPt [tid] = ptRaw;
  sFoc[tid] = -0.25f * om * om * __logf(ptCl);
  #pragma unroll
  for (int c = 0; c < 16; ++c)
    sMA[tid * 16 + c] = (_Float16)((c < NC) ? lg[c] * inv : 0.0f);   // probs matrix
  __syncthreads();

  // ---- per-wave WMMA reductions (row 0 = pt, row 1 = ones, row 2 = focal) ----
  const int  lane  = tid & 31;
  const int  wbase = tid & ~31;
  const int  m     = lane & 15;
  const bool lo    = lane < 16;

  v16h A;
  #pragma unroll
  for (int j = 0; j < 16; ++j) {
    int k = amapK(j, lo);
    float v = 0.0f;
    if      (m == 0) v = sPt [wbase + k];
    else if (m == 1) v = 1.0f;
    else if (m == 2) v = sFoc[wbase + k];
    A[j] = (_Float16)v;
  }

  // wmma #1: B = onehot(target) -> D[0]=inter, D[1]=tcount, D[2]=focal-per-class
  {
    v16h Bv;
    const int off = lo ? 0 : 16;
    #pragma unroll
    for (int j = 0; j < 16; ++j) {
      int t = sCt[wbase + off + j];
      Bv[j] = (_Float16)((m < NC && t == m) ? 1.0f : 0.0f);
    }
    v8f D = {};
    D = __builtin_amdgcn_wmma_f32_16x16x32_f16(false, A, false, Bv, (short)0, D, false, false);
    if (lo && m < NC) {
      atomicAdd(&sInter[m], D[0]);
      atomicAdd(&sTc[m],    D[1]);
      atomicAdd(&sFc[m],    D[2]);
    }
  }
  // wmma #2: B = probs -> D[1] = per-class prob sums
  {
    v16h Bv = gatherB(sMA, wbase, lane);
    v8f D = {};
    D = __builtin_amdgcn_wmma_f32_16x16x32_f16(false, A, false, Bv, (short)0, D, false, false);
    if (lo && m < NC) atomicAdd(&sPs[m], D[1]);
  }
  __syncthreads();   // all waves done reading sMA before edge pass overwrites it

  // ---- edge pass: 9 neighbors in registers, 10-class VALU loop ----
  {
    const int t00 = sT[ py      * 18 + px    ], t01 = sT[ py      * 18 + px + 1], t02 = sT[ py      * 18 + px + 2];
    const int t10 = sT[(py + 1) * 18 + px    ], t11 = tc,                         t12 = sT[(py + 1) * 18 + px + 2];
    const int t20 = sT[(py + 2) * 18 + px    ], t21 = sT[(py + 2) * 18 + px + 1], t22 = sT[(py + 2) * 18 + px + 2];

    sMA[tid * 16 + 0] = (_Float16)0.0f;
    sMB[tid * 16 + 0] = (_Float16)0.0f;
    #pragma unroll
    for (int c = 1; c < NC; ++c) {
      int cnt9 = (t00 == c) + (t01 == c) + (t02 == c)
               + (t10 == c) + (t11 == c) + (t12 == c)
               + (t20 == c) + (t21 == c) + (t22 == c);
      float dval = 0.f, nval = 0.f;
      if (cnt9 != 0 && cnt9 != 9) {                  // boundary
        dval = 1.f;
        const int tm = (t11 == c), pm = (pred == c);
        if (tm != pm) {
          if (tm) nval = 1.0f;                        // dist = 0
          else {
            int n4 = (t01 == c) + (t21 == c) + (t10 == c) + (t12 == c);
            nval = n4 ? 0.36787944117144233f          // exp(-1)
                      : 0.24311673443421421f;         // exp(-sqrt(2))
          }
        }
      }
      sMA[tid * 16 + c] = (_Float16)dval;             // boundary indicator
      sMB[tid * 16 + c] = (_Float16)nval;             // weighted mismatch
    }
    #pragma unroll
    for (int c = NC; c < 16; ++c) {
      sMA[tid * 16 + c] = (_Float16)0.0f;
      sMB[tid * 16 + c] = (_Float16)0.0f;
    }
  }
  __syncthreads();

  // wmma #3: B = boundary -> D[1] = den ; wmma #4: B = weights -> D[1] = num
  {
    v16h Bv = gatherB(sMA, wbase, lane);
    v8f D = {};
    D = __builtin_amdgcn_wmma_f32_16x16x32_f16(false, A, false, Bv, (short)0, D, false, false);
    if (lo && m < NC) atomicAdd(&sDen[m], D[1]);
  }
  {
    v16h Bv = gatherB(sMB, wbase, lane);
    v8f D = {};
    D = __builtin_amdgcn_wmma_f32_16x16x32_f16(false, A, false, Bv, (short)0, D, false, false);
    if (lo && m < NC) atomicAdd(&sNum[m], D[1]);
  }
  __syncthreads();

  if (tid < NC) {
    atomicAdd(&acc->inter[b][tid], sInter[tid]);
    atomicAdd(&acc->tcnt [b][tid], sTc[tid]);
    atomicAdd(&acc->focC [b][tid], sFc[tid]);
    atomicAdd(&acc->psum [b][tid], sPs[tid]);
    atomicAdd(&acc->num  [b][tid], sNum[tid]);
    atomicAdd(&acc->den  [b][tid], sDen[tid]);
  }
}

__global__ __launch_bounds__(128) void loss_final(const Accum* __restrict__ acc,
                                                  float* __restrict__ out)
{
  __shared__ float sd[128], sf[128], se2[128];
  const int t = threadIdx.x;
  float d = 0.f, f = 0.f, e = 0.f;
  if (t < NB * NC) {
    int b = t / NC, c = t % NC;
    float card = acc->psum[b][c] + acc->tcnt[b][c] + 1e-6f;
    d = (2.0f * acc->inter[b][c] + 1e-6f) / card;
    f = acc->focC[b][c];
  }
  if (t < NB * 10) {
    int b = t / 10, c = t % 10 + 1;
    float den = acc->den[b][c];
    e = (den > 0.f) ? acc->num[b][c] / fmaxf(den, 1.0f) : 0.f;
  }
  sd[t] = d; sf[t] = f; se2[t] = e;
  __syncthreads();
  if (t == 0) {
    float D = 0.f, F = 0.f, E = 0.f;
    for (int i = 0; i < 128; ++i) { D += sd[i]; F += sf[i]; E += se2[i]; }
    float dice_loss  = 1.0f - D / (float)(NB * NC);
    float focal_loss = F / ((float)NB * IH * IW);
    float edge_loss  = E / (float)(NB * 10);
    out[0] = dice_loss + focal_loss + edge_loss;
    out[1] = dice_loss;
    out[2] = focal_loss;
    out[3] = edge_loss;
  }
}

extern "C" void kernel_launch(void* const* d_in, const int* in_sizes, int n_in,
                              void* d_out, int out_size, void* d_ws, size_t ws_size,
                              hipStream_t stream) {
  const float* logits = (const float*)d_in[0];
  const int*   tgt    = (const int*)d_in[1];
  float*       out    = (float*)d_out;
  Accum*       acc    = (Accum*)d_ws;

  const int nfl = (int)(sizeof(Accum) / sizeof(float));
  loss_init <<<(nfl + 255) / 256, 256, 0, stream>>>((float*)d_ws, nfl);
  loss_main <<<NB * (IH / TILE) * (IW / TILE), 256, 0, stream>>>(logits, tgt, acc);
  loss_final<<<1, 128, 0, stream>>>(acc, out);
}
